// HardBatchMiningTripletLoss_26731876451185
// MI455X (gfx1250) — compile-verified, matrix operand order checked by reading, not run on hardware
//
#include <hip/hip_runtime.h>
#include <math.h>

typedef __attribute__((ext_vector_type(16))) __bf16 v16bf;
typedef __attribute__((ext_vector_type(8)))  __bf16 v8bf;
typedef __attribute__((ext_vector_type(8)))  float  v8f;
typedef __attribute__((ext_vector_type(4)))  float  v4f;

#define B_N     4096
#define D_K     2048
#define SPAD    2056     // LDS row stride in bf16: 2048+8 -> stride%64 words == 4, bank-conflict-free
#define MARGIN  0.3f
#define ROWS_WG 32       // rows per workgroup (2 row tiles of 16)
#define NWAVE   8        // 256 threads

// ---------------------------------------------------------------------------
// Kernel 1: convert X (f32) -> bf16 workspace, compute exact f32 row norms,
// zero the output accumulator. One block per row.
// ---------------------------------------------------------------------------
__global__ __launch_bounds__(256) void prep_kernel(const float* __restrict__ x,
                                                   __bf16* __restrict__ xb,
                                                   float* __restrict__ sq,
                                                   float* __restrict__ out) {
  const int row = blockIdx.x;
  const int tid = threadIdx.x;
  const float* src = x + (size_t)row * D_K + tid * 8;
  v4f f0 = *(const v4f*)(src);
  v4f f1 = *(const v4f*)(src + 4);
  v8bf b;
  float ss = 0.f;
#pragma unroll
  for (int i = 0; i < 4; ++i) { b[i]     = (__bf16)f0[i]; ss += f0[i] * f0[i]; }
#pragma unroll
  for (int i = 0; i < 4; ++i) { b[4 + i] = (__bf16)f1[i]; ss += f1[i] * f1[i]; }
  *(v8bf*)(xb + (size_t)row * D_K + tid * 8) = b;

  // block reduction of sum of squares (8 wave32s)
#pragma unroll
  for (int m = 16; m >= 1; m >>= 1) ss += __shfl_xor(ss, m, 32);
  __shared__ float red[8];
  const int wave = tid >> 5;
  const int lane = tid & 31;
  if (lane == 0) red[wave] = ss;
  __syncthreads();
  if (tid == 0) {
    float s = 0.f;
#pragma unroll
    for (int w = 0; w < 8; ++w) s += red[w];
    sq[row] = s;
    if (row == 0) out[0] = 0.f;   // zero accumulator before kernel 2 (same stream)
  }
}

// ---------------------------------------------------------------------------
// Kernel 2: fused  dist = sqrt(||xi||^2 + ||xj||^2 - 2*WMMA(xi.xj))  +
// hardest positive/negative mining + margin loss.
// 128 workgroups x 256 threads; each workgroup owns a 32-row LDS panel of A;
// each wave computes a 2x2 grid of 16x16 WMMA tiles so every B fragment
// fetched from L2 is reused across two row tiles (halves L2 traffic).
// ---------------------------------------------------------------------------
__global__ __launch_bounds__(256) void triplet_kernel(const __bf16* __restrict__ xb,
                                                      const float* __restrict__ sq,
                                                      const int* __restrict__ tgt,
                                                      float* __restrict__ out) {
  __shared__ __bf16 As[ROWS_WG * SPAD];        // ~131.6 KB
  __shared__ float dposS[NWAVE][ROWS_WG];
  __shared__ float dnegS[NWAVE][ROWS_WG];
  __shared__ float lossS[ROWS_WG];

  const int tid     = threadIdx.x;
  const int wave    = tid >> 5;
  const int lane    = tid & 31;
  const int l15     = lane & 15;
  const int lhalf   = lane >> 4;          // 0: K 0..7/16..23 & rows 0..7 ; 1: K 8..15/24..31 & rows 8..15
  const int rowBase = blockIdx.x * ROWS_WG;

  // Stage the 32-row A panel (bf16) into padded LDS: 32 rows x 256 16B chunks
  for (int c = tid; c < ROWS_WG * (D_K / 8); c += 256) {
    const int r   = c >> 8;               // 256 chunks per row
    const int off = (c & 255) * 8;
    *(v8bf*)&As[r * SPAD + off] = *(const v8bf*)(xb + (size_t)(rowBase + r) * D_K + off);
  }
  __syncthreads();

  // Per-lane row metadata (C-layout: VGPR r <-> row r + 8*lhalf, col = l15)
  float sqr[2][8]; int trow[2][8];
#pragma unroll
  for (int rt = 0; rt < 2; ++rt)
#pragma unroll
    for (int r = 0; r < 8; ++r) {
      const int g = rowBase + rt * 16 + r + 8 * lhalf;
      sqr[rt][r]  = sq[g];
      trow[rt][r] = tgt[g];
    }

  float dpos[2][8], dneg[2][8];
#pragma unroll
  for (int rt = 0; rt < 2; ++rt)
#pragma unroll
    for (int r = 0; r < 8; ++r) { dpos[rt][r] = -__builtin_inff(); dneg[rt][r] = __builtin_inff(); }

  const v8f vzero = {0.f, 0.f, 0.f, 0.f, 0.f, 0.f, 0.f, 0.f};

  // 128 column groups of 32 columns (2 col tiles); 8 waves stride the groups
  for (int g = wave; g < B_N / 32; g += NWAVE) {
    const int col0 = g * 32;

    // Warm caches for the next group's B rows (global_prefetch_b8)
    const int gn = g + NWAVE;
    if (gn < B_N / 32)
      __builtin_prefetch(xb + (size_t)(gn * 32 + lane) * D_K, 0, 1);

    v8f acc[2][2];
#pragma unroll
    for (int rt = 0; rt < 2; ++rt)
#pragma unroll
      for (int ct = 0; ct < 2; ++ct) acc[rt][ct] = vzero;

#pragma unroll 2
    for (int k0 = 0; k0 < D_K; k0 += 32) {
      const int koff = k0 + lhalf * 8;

      v16bf a[2];
#pragma unroll
      for (int rt = 0; rt < 2; ++rt) {
        v8bf alo = *(const v8bf*)&As[(rt * 16 + l15) * SPAD + koff];
        v8bf ahi = *(const v8bf*)&As[(rt * 16 + l15) * SPAD + koff + 16];
        a[rt] = __builtin_shufflevector(alo, ahi,
                                        0, 1, 2, 3, 4, 5, 6, 7,
                                        8, 9, 10, 11, 12, 13, 14, 15);
      }
      v16bf bfr[2];
#pragma unroll
      for (int ct = 0; ct < 2; ++ct) {
        const __bf16* bp = xb + (size_t)(col0 + ct * 16 + l15) * D_K + koff;
        v8bf blo = *(const v8bf*)bp;
        v8bf bhi = *(const v8bf*)(bp + 16);
        bfr[ct] = __builtin_shufflevector(blo, bhi,
                                          0, 1, 2, 3, 4, 5, 6, 7,
                                          8, 9, 10, 11, 12, 13, 14, 15);
      }
#pragma unroll
      for (int rt = 0; rt < 2; ++rt)
#pragma unroll
        for (int ct = 0; ct < 2; ++ct)
          acc[rt][ct] = __builtin_amdgcn_wmma_f32_16x16x32_bf16(
              /*neg_a=*/false, a[rt], /*neg_b=*/false, bfr[ct],
              /*c_mod=*/(short)0, acc[rt][ct], /*reuse_a=*/false, /*reuse_b=*/false);
    }

    // Epilogue: distances + hardest-pos/neg mining for the 2x2 tiles
#pragma unroll
    for (int ct = 0; ct < 2; ++ct) {
      const int col   = col0 + ct * 16 + l15;
      const float sqc = sq[col];
      const int   tc  = tgt[col];
#pragma unroll
      for (int rt = 0; rt < 2; ++rt)
#pragma unroll
        for (int r = 0; r < 8; ++r) {
          const float d2 = sqr[rt][r] + sqc - 2.0f * acc[rt][ct][r];
          const float d  = sqrtf(fmaxf(d2, 1e-12f));
          const bool same = (tc == trow[rt][r]);
          dpos[rt][r] = same ? fmaxf(dpos[rt][r], d) : dpos[rt][r];
          dneg[rt][r] = same ? dneg[rt][r] : fminf(dneg[rt][r], d);
        }
    }
  }

  // Reduce across the 16 column lanes of each half-wave (xor stays within halves)
#pragma unroll
  for (int m = 1; m <= 8; m <<= 1) {
#pragma unroll
    for (int rt = 0; rt < 2; ++rt)
#pragma unroll
      for (int r = 0; r < 8; ++r) {
        dpos[rt][r] = fmaxf(dpos[rt][r], __shfl_xor(dpos[rt][r], m, 32));
        dneg[rt][r] = fminf(dneg[rt][r], __shfl_xor(dneg[rt][r], m, 32));
      }
  }
  if (l15 == 0) {
#pragma unroll
    for (int rt = 0; rt < 2; ++rt)
#pragma unroll
      for (int r = 0; r < 8; ++r) {
        dposS[wave][rt * 16 + 8 * lhalf + r] = dpos[rt][r];
        dnegS[wave][rt * 16 + 8 * lhalf + r] = dneg[rt][r];
      }
  }
  __syncthreads();

  // Combine the 8 waves, compute per-row margin loss, accumulate to out
  if (tid < ROWS_WG) {
    float dp = dposS[0][tid], dn = dnegS[0][tid];
#pragma unroll
    for (int w = 1; w < NWAVE; ++w) {
      dp = fmaxf(dp, dposS[w][tid]);
      dn = fminf(dn, dnegS[w][tid]);
    }
    lossS[tid] = fmaxf(dp - dn + MARGIN, 0.0f);
  }
  __syncthreads();
  if (tid == 0) {
    float s = 0.f;
#pragma unroll
    for (int i = 0; i < ROWS_WG; ++i) s += lossS[i];
    atomicAdd(out, s * (1.0f / B_N));
  }
}

// ---------------------------------------------------------------------------
extern "C" void kernel_launch(void* const* d_in, const int* in_sizes, int n_in,
                              void* d_out, int out_size, void* d_ws, size_t ws_size,
                              hipStream_t stream) {
  const float* x   = (const float*)d_in[0];
  const int*   tgt = (const int*)d_in[1];
  float*       out = (float*)d_out;

  __bf16* xb = (__bf16*)d_ws;                                              // 16.8 MB bf16 copy of X
  float*  sq = (float*)((char*)d_ws + (size_t)B_N * D_K * sizeof(__bf16)); // 16 KB row norms

  prep_kernel<<<B_N, 256, 0, stream>>>(x, xb, sq, out);
  triplet_kernel<<<B_N / ROWS_WG, 256, 0, stream>>>(xb, sq, tgt, out);
}
